// Attn_28913719837141
// MI455X (gfx1250) — compile-verified, matrix operand order checked by reading, not run on hardware
//
#include <hip/hip_runtime.h>
#include <hip/hip_bf16.h>

typedef unsigned short u16;
typedef unsigned int   u32;
typedef unsigned long long u64;
typedef __attribute__((ext_vector_type(16))) __bf16 v16bf;
typedef __attribute__((ext_vector_type(8)))  float  v8f;
typedef __attribute__((ext_vector_type(4)))  unsigned int v4u;
typedef __attribute__((ext_vector_type(8)))  int    v8i;
typedef __attribute__((ext_vector_type(4)))  int    v4i;

#define DEV static __device__ __forceinline__

// ---------- scalar bf16 helpers (RNE) ----------
DEV u16 f2bf(float f) {
    u32 u = __builtin_bit_cast(u32, f);
    u32 r = u + 0x7FFFu + ((u >> 16) & 1u);
    return (u16)(r >> 16);
}
DEV float bf2f(u16 v) {
    u32 u = ((u32)v) << 16;
    return __builtin_bit_cast(float, u);
}

// ---------- WMMA fragment helpers ----------
// CDNA5 16-bit A/B fragment layout (16x32 / 32x16):
//   lanes 0-15  : K = {0..7, 16..23}, lanes 16-31 : K = {8..15, 24..31}
// One fragment per lane = two aligned 16-byte loads from a K-contiguous row.
union Frag { uint4 u4[2]; v16bf v; };

DEV v16bf load_frag32(const u16* rowk0, int lane) {
    const int hi8 = (lane & 16) ? 8 : 0;
    Frag f;
    f.u4[0] = *(const uint4*)(rowk0 + hi8);
    f.u4[1] = *(const uint4*)(rowk0 + 16 + hi8);
    return f.v;
}

DEV v8f wmma_bf16(v16bf a, v16bf b, v8f c) {
    return __builtin_amdgcn_wmma_f32_16x16x32_bf16(false, a, false, b, (short)0, c,
                                                   false, false);
}

// ---------- TDM: async 2-D tile load Global->LDS ----------
// D# packing per CDNA5 ISA (2-D tensor, groups 2/3 zero).
DEV void tdm_load_tile_bf16(u32 lds_addr, u64 gaddr,
                            u32 tensor_dim0, u32 tensor_dim1,
                            u32 tile_dim0, u32 tile_dim1, u32 stride0)
{
    v4u g0 = { 1u /*count=1*/,
               lds_addr,
               (u32)gaddr,
               (u32)((gaddr >> 32) & 0x01FFFFFFu) | (2u << 30) /*type=2*/ };
    v8i g1 = { (int)(1u << 16),                       // data_size=1 (2 bytes)
               (int)((tensor_dim0 & 0xFFFFu) << 16),  // tensor_dim0[15:0] @ 63:48
               (int)((tensor_dim0 >> 16) | ((tensor_dim1 & 0xFFFFu) << 16)),
               (int)((tensor_dim1 >> 16) | (tile_dim0 << 16)),
               (int)(tile_dim1),                      // tile_dim1, tile_dim2=0
               (int)(stride0),                        // tensor_dim0_stride[31:0]
               0, 0 };
    v4i z4 = {0, 0, 0, 0};
#if defined(__clang_major__) && (__clang_major__ >= 23)
    v8i z8 = {0, 0, 0, 0, 0, 0, 0, 0};
    __builtin_amdgcn_tensor_load_to_lds(g0, g1, z4, z4, z8, 0);
#else
    __builtin_amdgcn_tensor_load_to_lds(g0, g1, z4, z4, 0);
#endif
}

// ---------- fp32 -> bf16 convert ----------
__global__ __launch_bounds__(256)
void f2bf_kernel(const float* __restrict__ in, u16* __restrict__ out, int n) {
    int i = blockIdx.x * 256 + threadIdx.x;
    if (i < n) out[i] = f2bf(in[i]);
}

// ---------- tiled bf16 WMMA GEMM: C[MxN] = A[MxK] * Bw[KxN] + bias ----------
// A tile staged by TDM (row-major, contiguous in LDS); B tile staged manually
// with transpose to [n][k] so fragments are two aligned ds_load_b128.
template<bool OUT_BF16>
__global__ __launch_bounds__(256)
void gemm_bf16_kernel(const u16* __restrict__ A, const u16* __restrict__ Bw,
                      const float* __restrict__ bias, void* __restrict__ Cout,
                      int M, int N, int K)
{
    __shared__ __align__(16) u16 As[128 * 32];  // [m][k] (TDM dest)
    __shared__ __align__(16) u16 Bs[64 * 32];   // [n][k]

    const int tid  = threadIdx.x;
    const int lane = tid & 31;
    const int wv   = tid >> 5;
    const int wm   = wv & 3, wn = wv >> 2;
    const long m0  = (long)blockIdx.x * 128;
    const long n0  = (long)blockIdx.y * 64;

    v8f acc[2][2] = {};

    const int bidx = tid * 8;
    const int bk = bidx >> 6, bc = bidx & 63;
    const u32 ldsAs = (u32)(uintptr_t)(void*)&As[0];
    const u64 abase = (u64)(uintptr_t)A + (u64)(m0 * (long)K) * 2;

    for (int k0 = 0; k0 < K; k0 += 32) {
        // A tile 128x32 via Tensor Data Mover (issued once per block)
        if (wv == 0)
            tdm_load_tile_bf16(ldsAs, abase + (u64)k0 * 2,
                               (u32)K, (u32)M, 32, 128, (u32)K);
        // B tile 32x64 staged transposed while the DMA runs
        uint4 d = *(const uint4*)(Bw + (long)(k0 + bk) * N + n0 + bc);
        const u16* dp = (const u16*)&d;
#pragma unroll
        for (int j = 0; j < 8; ++j) Bs[(bc + j) * 32 + bk] = dp[j];
        if (k0 + 32 < K)
            __builtin_prefetch(Bw + (long)(k0 + 32 + bk) * N + n0 + bc, 0, 0);

        if (wv == 0) __builtin_amdgcn_s_wait_tensorcnt(0);
        __syncthreads();

        v16bf af[2], bfr[2];
#pragma unroll
        for (int mi = 0; mi < 2; ++mi)
            af[mi] = load_frag32(&As[(wm * 32 + mi * 16 + (lane & 15)) * 32], lane);
#pragma unroll
        for (int ni = 0; ni < 2; ++ni)
            bfr[ni] = load_frag32(&Bs[(wn * 32 + ni * 16 + (lane & 15)) * 32], lane);
#pragma unroll
        for (int mi = 0; mi < 2; ++mi)
#pragma unroll
            for (int ni = 0; ni < 2; ++ni)
                acc[mi][ni] = wmma_bf16(af[mi], bfr[ni], acc[mi][ni]);
        __syncthreads();
    }

    const int rsel = (lane & 16) ? 8 : 0;
    const int cl = lane & 15;
#pragma unroll
    for (int mi = 0; mi < 2; ++mi)
#pragma unroll
        for (int ni = 0; ni < 2; ++ni) {
            long col = n0 + wn * 32 + ni * 16 + cl;
            float bv = bias[col];
#pragma unroll
            for (int vg = 0; vg < 8; ++vg) {
                long row = m0 + wm * 32 + mi * 16 + rsel + vg;
                float val = acc[mi][ni][vg] + bv;
                if (OUT_BF16) ((u16*)Cout)[row * N + col] = f2bf(val);
                else          ((float*)Cout)[row * N + col] = val;
            }
        }
}

// ---------- RMS-norm q,k + head relayout: qkv[B,S,3,H,64] -> Qn/Kn/Vt[B,H,S,64] ----------
// The attention scale 1/sqrt(64) is folded into Q here (scores are linear in Q),
// removing 16 VALU muls per key-block from the flash hot loop.
__global__ __launch_bounds__(256)
void rmsnorm_split_kernel(const u16* __restrict__ QKV,
                          const float* __restrict__ qs, const float* __restrict__ ks,
                          u16* __restrict__ Qn, u16* __restrict__ Kn, u16* __restrict__ Vt)
{
    const int lane = threadIdx.x & 31;
    const int wv   = threadIdx.x >> 5;
    const long row = (long)blockIdx.x * 8 + wv;
    const int  h   = (int)(row & 15);
    const long bs  = row >> 4;
    const long b   = bs >> 11;
    const long s   = bs & 2047;
    const int  d   = lane * 2;
    const long dst = (((b * 16 + h) * 2048) + s) * 64 + d;

    const u16* q = QKV + bs * 3072 + h * 64 + d;
    const u16* k = q + 1024;
    const u16* v = q + 2048;

    *(u32*)(Vt + dst) = *(const u32*)v;

    {
        u32 pk = *(const u32*)q;
        float f0 = bf2f((u16)pk), f1 = bf2f((u16)(pk >> 16));
        float ss = f0 * f0 + f1 * f1;
#pragma unroll
        for (int off = 16; off >= 1; off >>= 1) ss += __shfl_xor(ss, off, 32);
        float inv = rsqrtf(ss * (1.0f / 64.0f) + 1e-6f) * 0.125f;  // folds 1/sqrt(hd)
        u16 o0 = f2bf(f0 * inv * qs[d]);
        u16 o1 = f2bf(f1 * inv * qs[d + 1]);
        *(u32*)(Qn + dst) = (u32)o0 | ((u32)o1 << 16);
    }
    {
        u32 pk = *(const u32*)k;
        float f0 = bf2f((u16)pk), f1 = bf2f((u16)(pk >> 16));
        float ss = f0 * f0 + f1 * f1;
#pragma unroll
        for (int off = 16; off >= 1; off >>= 1) ss += __shfl_xor(ss, off, 32);
        float inv = rsqrtf(ss * (1.0f / 64.0f) + 1e-6f);
        u16 o0 = f2bf(f0 * inv * ks[d]);
        u16 o1 = f2bf(f1 * inv * ks[d + 1]);
        *(u32*)(Kn + dst) = (u32)o0 | ((u32)o1 << 16);
    }
}

// ---------- flash attention ----------
// Block = 128 threads (4 waves), 64 query rows of one (b,h).
// K tiles double-buffered via TDM (DMA for tile j+32 overlaps compute on j).
// V tile staged transposed [n][k]; row-sums of P computed with a WMMA against
// an all-ones B fragment (every column of the D tile = row sum -> no shuffles).
// Q is pre-scaled by 1/sqrt(hd) in the rmsnorm kernel.
__global__ __launch_bounds__(128)
void flash_attn_kernel(const u16* __restrict__ Q, const u16* __restrict__ Kd,
                       const u16* __restrict__ V, u16* __restrict__ O)
{
    __shared__ __align__(16) u16 Ks[2][32 * 64];    // double-buffered TDM dest
    __shared__ __align__(16) u16 Vs[64 * 32];       // [hd][key] transposed
    __shared__ __align__(16) u16 Plds[4][16 * 32];  // per-wave P bounce

    const int tid  = threadIdx.x;
    const int lane = tid & 31;
    const int wv   = tid >> 5;
    const int bh   = blockIdx.x >> 5;   // B*H = 32
    const int qblk = blockIdx.x & 31;   // S/64 = 32
    const int h    = bh & 15, b = bh >> 4;
    const long base = (long)bh * 2048 * 64;
    const int l15  = lane & 15;
    const int rsel = (lane & 16) ? 8 : 0;

    const int q0 = qblk * 64 + wv * 16;
    const u16* qrow = Q + base + (long)(q0 + l15) * 64;
    v16bf aq0 = load_frag32(qrow, lane);
    v16bf aq1 = load_frag32(qrow + 32, lane);

    v16bf ones;
#pragma unroll
    for (int i = 0; i < 16; ++i) ones[i] = (__bf16)1.0f;

    v8f ao[4] = {};
    float mrow[8], lrow[8];
#pragma unroll
    for (int i = 0; i < 8; ++i) { mrow[i] = -1e30f; lrow[i] = 0.f; }

    u16* pbuf = &Plds[wv][0];
    const u32 ldsKs[2] = { (u32)(uintptr_t)(void*)&Ks[0][0],
                           (u32)(uintptr_t)(void*)&Ks[1][0] };
    const u64 kbase = (u64)(uintptr_t)Kd + (u64)base * 2;

    const int vrow  = tid >> 2;        // 0..31 : key row of V tile
    const int vcol0 = (tid & 3) * 16;  // head-dim column group

    // prologue: fill buffer 0
    if (wv == 0)
        tdm_load_tile_bf16(ldsKs[0], kbase, 64, 65536, 64, 32, 64);

    int buf = 0;
    for (int j = 0; j < 2048; j += 32, buf ^= 1) {
        // --- stage V tile transposed (previous iter's readers done: end barrier) ---
        {
            const u16* vsrc = V + base + (long)(j + vrow) * 64 + vcol0;
            uint4 d0 = *(const uint4*)vsrc;
            uint4 d1 = *(const uint4*)(vsrc + 8);
            const u16* p0 = (const u16*)&d0;
            const u16* p1 = (const u16*)&d1;
#pragma unroll
            for (int i = 0; i < 8; ++i) {
                Vs[(vcol0 + i) * 32 + vrow]     = p0[i];
                Vs[(vcol0 + 8 + i) * 32 + vrow] = p1[i];
            }
        }
        if (wv == 0) __builtin_amdgcn_s_wait_tensorcnt(0);  // K tile `buf` landed
        __syncthreads();
        // pipeline: kick DMA for the next K tile into the other buffer
        if (wv == 0 && j + 32 < 2048)
            tdm_load_tile_bf16(ldsKs[buf ^ 1], kbase + (u64)(j + 32) * 128,
                               64, 65536, 64, 32, 64);

        // --- scores 16x32 = Qs(16x64) @ K^T (scale pre-folded into Q) ---
        const u16* kt = &Ks[buf][0];
        v8f sc[2];
#pragma unroll
        for (int nh = 0; nh < 2; ++nh) {
            const u16* krow = kt + (nh * 16 + l15) * 64;
            v8f t = {};
            t = wmma_bf16(aq0, load_frag32(krow, lane), t);
            t = wmma_bf16(aq1, load_frag32(krow + 32, lane), t);
            sc[nh] = t;
        }

        // --- online softmax: only the row-max needs a butterfly ---
        float alpha[8];
#pragma unroll
        for (int vg = 0; vg < 8; ++vg) {
            float s0 = sc[0][vg];
            float s1 = sc[1][vg];
            float mloc = fmaxf(s0, s1);
#pragma unroll
            for (int off = 8; off >= 1; off >>= 1)
                mloc = fmaxf(mloc, __shfl_xor(mloc, off, 32));
            float mnew = fmaxf(mrow[vg], mloc);
            alpha[vg] = __expf(mrow[vg] - mnew);
            mrow[vg] = mnew;
            float p0 = __expf(s0 - mnew);
            float p1 = __expf(s1 - mnew);
            pbuf[(vg + rsel) * 32 + l15]      = f2bf(p0);
            pbuf[(vg + rsel) * 32 + 16 + l15] = f2bf(p1);
        }
#pragma unroll
        for (int nb = 0; nb < 4; ++nb)
#pragma unroll
            for (int vg = 0; vg < 8; ++vg)
                ao[nb][vg] *= alpha[vg];

        // --- P fragment from per-wave LDS bounce ---
        v16bf ap = load_frag32(&pbuf[l15 * 32], lane);

        // row sums of P via matrix pipe: every column of (P @ ones) = row sum
        v8f psum = wmma_bf16(ap, ones, v8f{});
#pragma unroll
        for (int vg = 0; vg < 8; ++vg)
            lrow[vg] = lrow[vg] * alpha[vg] + psum[vg];

        // --- O += P(16x32) @ V(32x64) ---
#pragma unroll
        for (int nb = 0; nb < 4; ++nb)
            ao[nb] = wmma_bf16(ap, load_frag32(&Vs[(nb * 16 + l15) * 32], lane),
                               ao[nb]);
        __syncthreads();   // protect Vs (and Ks[buf]) before next staging
    }

    // --- normalize, write o to [B,S,D] ---
#pragma unroll
    for (int nb = 0; nb < 4; ++nb) {
#pragma unroll
        for (int vg = 0; vg < 8; ++vg) {
            float val = ao[nb][vg] / lrow[vg];
            long srow = (long)b * 2048 + q0 + rsel + vg;
            O[srow * 1024 + h * 64 + nb * 16 + l15] = f2bf(val);
        }
    }
}

// ---------- host glue ----------
extern "C" void kernel_launch(void* const* d_in, const int* in_sizes, int n_in,
                              void* d_out, int out_size, void* d_ws, size_t ws_size,
                              hipStream_t stream)
{
    const float* x    = (const float*)d_in[0];
    const float* Wqkv = (const float*)d_in[1];
    const float* bqkv = (const float*)d_in[2];
    const float* Wout = (const float*)d_in[3];
    const float* bout = (const float*)d_in[4];
    const float* qs   = (const float*)d_in[5];
    const float* ks   = (const float*)d_in[6];

    char* p = (char*)d_ws;
    u16* Xb    = (u16*)p; p += (size_t)4194304  * 2;
    u16* Wqkvb = (u16*)p; p += (size_t)3145728  * 2;
    u16* Woutb = (u16*)p; p += (size_t)1048576  * 2;
    u16* QKVb  = (u16*)p; p += (size_t)12582912 * 2;
    u16* Qn    = (u16*)p; p += (size_t)4194304  * 2;
    u16* Kn    = (u16*)p; p += (size_t)4194304  * 2;
    u16* Vt    = (u16*)p; p += (size_t)4194304  * 2;
    u16* Ob    = (u16*)p; p += (size_t)4194304  * 2;

    f2bf_kernel<<<4194304 / 256, 256, 0, stream>>>(x,    Xb,    4194304);
    f2bf_kernel<<<3145728 / 256, 256, 0, stream>>>(Wqkv, Wqkvb, 3145728);
    f2bf_kernel<<<1048576 / 256, 256, 0, stream>>>(Wout, Woutb, 1048576);

    gemm_bf16_kernel<true><<<dim3(4096 / 128, 3072 / 64), 256, 0, stream>>>(
        Xb, Wqkvb, bqkv, (void*)QKVb, 4096, 3072, 1024);

    rmsnorm_split_kernel<<<65536 / 8, 256, 0, stream>>>(QKVb, qs, ks, Qn, Kn, Vt);

    flash_attn_kernel<<<1024, 128, 0, stream>>>(Qn, Kn, Vt, Ob);

    gemm_bf16_kernel<false><<<dim3(4096 / 128, 1024 / 64), 256, 0, stream>>>(
        Ob, Woutb, bout, d_out, 4096, 1024, 1024);
}
